// ConvStyled3d_unet_51384988729725
// MI455X (gfx1250) — matrix operand order repaired and checked by reading, not compiled, not run on hardware
//
#include <hip/hip_runtime.h>
#include <hip/hip_bf16.h>

// ---------------- types for WMMA ----------------
typedef __attribute__((ext_vector_type(16))) __bf16 v16bf;
typedef __attribute__((ext_vector_type(8)))  float  v8f;

struct __attribute__((aligned(16))) U4 { unsigned int a, b, c, d; };
struct U8 { U4 lo, hi; };

// ---------------- problem constants ----------------
constexpr int NB   = 4;      // batch
constexpr int CIN  = 64;
constexpr int COUT = 64;
constexpr int SSZ  = 512;    // style size
constexpr int DD   = 64;     // input spatial
constexpr int KK3  = 3;      // kernel
constexpr int OD   = DD - KK3 + 1;          // 62
constexpr int PT   = OD * OD * OD;          // 238328 output positions
constexpr int KTOT = CIN * KK3 * KK3 * KK3; // 1728 (K order: kk*64 + cin)
constexpr int KCH  = KTOT / 32;             // 54 K-chunks of 32
constexpr int WSP  = KTOT + 8;              // padded weight LDS row (1736 elem)
constexpr int ASP  = 40;                    // padded activation LDS row (elems)
constexpr int MT   = 256;                   // spatial tile per workgroup
constexpr int BUFB = MT * ASP * 2;          // bytes per activation buffer (20480)

// fp32 -> bf16 round-to-nearest-even
static __device__ __forceinline__ unsigned int f2bf(float f) {
    unsigned int u = __builtin_bit_cast(unsigned int, f);
    unsigned int r = u + 0x7FFFu + ((u >> 16) & 1u);
    return r >> 16;
}

static __device__ __forceinline__ v16bf load_frag(const unsigned short* p0,
                                                  const unsigned short* p1) {
    U8 raw;
    raw.lo = *reinterpret_cast<const U4*>(p0);
    raw.hi = *reinterpret_cast<const U4*>(p1);
    return __builtin_bit_cast(v16bf, raw);
}

static __device__ __forceinline__ void wait_async0() {
#if __has_builtin(__builtin_amdgcn_s_wait_asynccnt)
    __builtin_amdgcn_s_wait_asynccnt(0);
#else
    asm volatile("s_wait_asynccnt 0x0" ::: "memory");
#endif
}

// async copy 64 contiguous bytes (per lane) global -> LDS
// ISA 15.18.3: offset is added to both the LDS address (VDST) and global address (SADDR + VADDR)
static __device__ __forceinline__ void async_copy64(unsigned ldsAddr, unsigned goff,
                                                    const unsigned short* sbase) {
    asm volatile(
        "global_load_async_to_lds_b128 %0, %1, %2 offset:0\n\t"
        "global_load_async_to_lds_b128 %0, %1, %2 offset:16\n\t"
        "global_load_async_to_lds_b128 %0, %1, %2 offset:32\n\t"
        "global_load_async_to_lds_b128 %0, %1, %2 offset:48"
        :: "v"(ldsAddr), "v"(goff), "s"(sbase) : "memory");
}

// ---------------- kernel 0: x (f32 NCDHW) -> xT (bf16 N,D,H,W,C) ----------------
__global__ void k_xform(const float* __restrict__ x, unsigned short* __restrict__ xT) {
    __shared__ unsigned short t[64][65];
    const int zy = blockIdx.x;          // z*64 + y
    const int n  = blockIdx.y;
    const int z  = zy >> 6, y = zy & 63;
    const int wi = threadIdx.x & 63;
    const int g4 = threadIdx.x >> 6;    // 0..3
    #pragma unroll 4
    for (int cc = 0; cc < 64; cc += 4) {
        int cin = cc + g4;
        float v = x[((((size_t)(n * CIN + cin) * DD + z) * DD + y) * DD) + wi];
        t[wi][cin] = (unsigned short)f2bf(v);
    }
    __syncthreads();
    #pragma unroll 4
    for (int ww = 0; ww < 64; ww += 4) {
        int w = ww + g4;
        xT[((((size_t)(n * DD + z) * DD + y) * DD + w) * CIN) + wi] = t[w][wi];
    }
}

// ---------------- kernel 1: st = s @ style_weight^T + style_bias ----------------
__global__ void k_style(const float* __restrict__ s,
                        const float* __restrict__ sw,
                        const float* __restrict__ sb,
                        float* __restrict__ stF) {
    int t = threadIdx.x;           // 256 threads: n = t/64, cin = t%64
    int n = t >> 6, cin = t & 63;
    float acc = sb[cin];
    const float* sr = s + n * SSZ;
    const float* wr = sw + cin * SSZ;
    for (int j = 0; j < SSZ; ++j) acc += sr[j] * wr[j];
    stF[t] = acc;
}

// ---------------- kernel 2: modulate + demodulate, emit bf16 weights ----------------
// output layout: wB[n][cout][k], k = kk*64 + cin  (K contiguous per cout)
__global__ void k_modw(const float* __restrict__ weight,
                       const float* __restrict__ stF,
                       unsigned short* __restrict__ wB) {
    int cout = blockIdx.x, n = blockIdx.y, tid = threadIdx.x;
    __shared__ float red[256];
    float vals[7];
    int   cnt = 0;
    float ss = 0.f;
    for (int e = tid; e < KTOT; e += 256) {
        int cin = e & 63, kk = e >> 6;
        float wv = weight[(cout * CIN + cin) * 27 + kk] * stF[n * CIN + cin];
        vals[cnt++] = wv;
        ss += wv * wv;
    }
    red[tid] = ss;
    __syncthreads();
    for (int st = 128; st > 0; st >>= 1) {
        if (tid < st) red[tid] += red[tid + st];
        __syncthreads();
    }
    float inv = rsqrtf(red[0] + 1e-8f);
    unsigned short* dst = wB + ((size_t)(n * COUT + cout)) * KTOT;
    cnt = 0;
    for (int e = tid; e < KTOT; e += 256)
        dst[e] = (unsigned short)f2bf(vals[cnt++] * inv);
}

// ---------------- kernel 3: implicit-GEMM conv via bf16 WMMA + async staging ----
// grid: x = spatial tiles (256 positions), y = cout half (32 couts), z = n
// block: 256 threads = 8 waves; wave w owns spatial rows w*32 .. w*32+31
__global__ void k_conv(const unsigned short* __restrict__ xT,
                       const unsigned short* __restrict__ wB,
                       const float* __restrict__ bias,
                       float* __restrict__ out) {
    extern __shared__ unsigned short smem[];
    unsigned short* wS = smem;                  // 32 couts * WSP
    unsigned short* aS = smem + 32 * WSP;       // 2 buffers of MT * ASP

    const int tid  = threadIdx.x;
    const int lane = tid & 31;
    const int wave = tid >> 5;
    const int n    = blockIdx.z;
    const int coutBase = blockIdx.y * 32;
    const long pBase   = (long)blockIdx.x * MT;

    // --- staging geometry: thread owns one spatial row m = tid, 32 cins/chunk ---
    int pc = (int)pBase + tid;
    if (pc > PT - 1) pc = PT - 1;               // clamp: garbage rows masked at store
    const int od = pc / (OD * OD);
    const int r0 = pc - od * (OD * OD);
    const int oh = r0 / OD;
    const int ow = r0 - oh * OD;
    // per-lane byte offset into xT[n] (channel-last), SGPR base carries n
    const unsigned xoff = (unsigned)((((od * DD + oh) * DD + ow) * CIN) * 2);
    const unsigned short* xTn = xT + (size_t)n * DD * DD * DD * CIN;
    const unsigned myDst = (unsigned)(uintptr_t)(aS) + tid * ASP * 2;

    // tap counters for the NEXT chunk to load (start at chunk 0)
    int ld_kd = 0, ld_kh = 0, ld_kw = 0, ld_half = 0;
    // prologue: async-load chunk 0 into buffer 0 (overlaps weight staging below)
    async_copy64(myDst, xoff, xTn);
    ld_half = 1;

    // --- stage 32 couts of bf16 weights into LDS (one time, padded rows) ---
    {
        const U4* src = reinterpret_cast<const U4*>(
            wB + ((size_t)(n * COUT + coutBase)) * KTOT);
        for (int r = 0; r < 32; ++r) {
            U4* dst = reinterpret_cast<U4*>(wS + r * WSP);
            const U4* sr = src + r * (KTOT / 8);
            for (int i = tid; i < KTOT / 8; i += 256) dst[i] = sr[i];
        }
    }

    v8f acc00 = {0.f,0.f,0.f,0.f,0.f,0.f,0.f,0.f};
    v8f acc01 = {0.f,0.f,0.f,0.f,0.f,0.f,0.f,0.f};
    v8f acc10 = {0.f,0.f,0.f,0.f,0.f,0.f,0.f,0.f};
    v8f acc11 = {0.f,0.f,0.f,0.f,0.f,0.f,0.f,0.f};

    const int arow = lane & 15;
    const int aoff = (lane & 16) ? 8 : 0;       // A-frag K start per ISA layout
    const int boff = (lane & 16) ? 8 : 0;       // B-frag K start per ISA layout
    const unsigned short* a0p = wS + arow * WSP + aoff;
    const unsigned short* a1p = wS + (16 + arow) * WSP + aoff;
    const unsigned short* bp0 = aS + (wave * 32 + (lane & 15)) * ASP + boff;
    const unsigned short* bp1 = bp0 + 16 * ASP;

    #pragma unroll 1          // keep accumulators in place (no acc reg rotation)
    for (int kc = 0; kc < KCH; ++kc) {
        wait_async0();                          // this wave's chunk-kc copies done
        __syncthreads();                        // everyone's done; prev reads done

        // async-prefetch chunk kc+1 into the other buffer (overlaps WMMAs)
        if (kc + 1 < KCH) {
            const unsigned cb =
                (unsigned)((((ld_kd * 4096 + ld_kh * 64 + ld_kw) * CIN) + ld_half * 32) * 2);
            const unsigned dbuf = ((kc + 1) & 1) ? BUFB : 0u;
            async_copy64(myDst + dbuf, xoff + cb, xTn);
            ld_half ^= 1;
            if (ld_half == 0) {
                if (++ld_kw == 3) { ld_kw = 0; if (++ld_kh == 3) { ld_kh = 0; ++ld_kd; } }
            }
        }

        const int kbase  = kc * 32;
        const int bufOfs = (kc & 1) ? (MT * ASP) : 0;   // elements
        v16bf a0 = load_frag(a0p + kbase, a0p + kbase + 16);
        v16bf a1 = load_frag(a1p + kbase, a1p + kbase + 16);
        v16bf b0 = load_frag(bp0 + bufOfs, bp0 + bufOfs + 16);
        v16bf b1 = load_frag(bp1 + bufOfs, bp1 + bufOfs + 16);

        acc00 = __builtin_amdgcn_wmma_f32_16x16x32_bf16(
                    false, a0, false, b0, (short)0, acc00, false, false);
        acc10 = __builtin_amdgcn_wmma_f32_16x16x32_bf16(
                    false, a1, false, b0, (short)0, acc10, false, false);
        acc01 = __builtin_amdgcn_wmma_f32_16x16x32_bf16(
                    false, a0, false, b1, (short)0, acc01, false, false);
        acc11 = __builtin_amdgcn_wmma_f32_16x16x32_bf16(
                    false, a1, false, b1, (short)0, acc11, false, false);
    }

    // --- store: lanes 0-15 write 16 consecutive positions (coalesced) ---
    const int mAdd = (lane & 16) ? 8 : 0;
    const long pcol0 = pBase + wave * 32 + (lane & 15);
    const long pcol1 = pcol0 + 16;
    if (pcol0 < PT) {
        #pragma unroll
        for (int r = 0; r < 8; ++r) {
            int co0 = coutBase + r + mAdd;
            int co1 = co0 + 16;
            out[(size_t)(n * COUT + co0) * PT + pcol0] = acc00[r] + bias[co0];
            out[(size_t)(n * COUT + co1) * PT + pcol0] = acc10[r] + bias[co1];
        }
    }
    if (pcol1 < PT) {
        #pragma unroll
        for (int r = 0; r < 8; ++r) {
            int co0 = coutBase + r + mAdd;
            int co1 = co0 + 16;
            out[(size_t)(n * COUT + co0) * PT + pcol1] = acc01[r] + bias[co0];
            out[(size_t)(n * COUT + co1) * PT + pcol1] = acc11[r] + bias[co1];
        }
    }
}

// ---------------- launcher ----------------
extern "C" void kernel_launch(void* const* d_in, const int* in_sizes, int n_in,
                              void* d_out, int out_size, void* d_ws, size_t ws_size,
                              hipStream_t stream) {
    const float* x   = (const float*)d_in[0];
    const float* s   = (const float*)d_in[1];
    const float* sw  = (const float*)d_in[2];
    const float* sb  = (const float*)d_in[3];
    const float* w   = (const float*)d_in[4];
    const float* bia = (const float*)d_in[5];
    float* out = (float*)d_out;

    float*          stF = (float*)d_ws;                                  // 256 f32
    unsigned short* wB  = (unsigned short*)((char*)d_ws + 1024);         // 442368 bf16
    unsigned short* xT  = (unsigned short*)((char*)d_ws + (1u << 20));   // 67.1M bf16

    k_xform<<<dim3(DD * DD, NB), 256, 0, stream>>>(x, xT);
    k_style<<<1, 256, 0, stream>>>(s, sw, sb, stF);
    k_modw<<<dim3(COUT, NB), 256, 0, stream>>>(w, stF, wB);

    const int tiles = (PT + MT - 1) / MT;                                // 931
    const size_t lds = (size_t)(32 * WSP + 2 * MT * ASP) * sizeof(unsigned short);
    k_conv<<<dim3(tiles, 2, NB), 256, lds, stream>>>(xT, wB, bia, out);
}
